// MultiLayerDirectedGraph_89584427860681
// MI455X (gfx1250) — matrix operand order repaired and checked by reading, not compiled, not run on hardware
//
#include <hip/hip_runtime.h>

// ---------------------------------------------------------------------------
// Types
// ---------------------------------------------------------------------------
typedef __attribute__((ext_vector_type(16))) __bf16 v16bf;
typedef __attribute__((ext_vector_type(8)))  __bf16 v8bf;
typedef __attribute__((ext_vector_type(8)))  float  v8f;
typedef __attribute__((ext_vector_type(4)))  float  v4f;

#define CAT16(lo, hi) __builtin_shufflevector(lo, hi, 0,1,2,3,4,5,6,7,8,9,10,11,12,13,14,15)

#define BQ 16
#define TQ 64
#define NN 50      // nodes per (b,t) incl audio
#define HD 512
#define MROWS (BQ * TQ * NN)   // 51200

// LDS row stride for 32-k tiles (keeps 16B alignment, breaks bank conflicts)
#define LSTR 40

// ---------------------------------------------------------------------------
// WMMA GEMM:  C[M x N] = epilogue( A[M x K](bf16) * Wb[N x K](bf16)^T + bias )
// mode 0: C = acc + bias
// mode 1: C = relu(acc + bias)
// mode 2: C = C + relu(acc + bias)
// mode 3: C = R + relu(acc)            (bias may be null)
// Block tile 128x128, 256 threads = 8 waves, each wave -> 32x64 output
// (8 v_wmma_f32_16x16x32_bf16 per K-step), double-buffered LDS.
// ---------------------------------------------------------------------------
__global__ __launch_bounds__(256)
void gemm_bf16(const __bf16* __restrict__ A, const __bf16* __restrict__ Wb,
               const float* __restrict__ bias, const float* __restrict__ R,
               float* __restrict__ C, int M, int N, int K, int mode)
{
    __shared__ __bf16 As[2][128 * LSTR];   // 128 rows x 32 k, double buffered
    __shared__ __bf16 Bs[2][128 * LSTR];   // 128 cols(n) x 32 k

    const int tid  = threadIdx.x;
    const int bm   = blockIdx.y * 128;
    const int bn   = blockIdx.x * 128;
    const int wid  = tid >> 5;
    const int lane = tid & 31;
    const int wm   = (wid & 3) * 32;   // wave M offset in tile (0,32,64,96)
    const int wn   = (wid >> 2) * 64;  // wave N offset in tile (0,64)
    const int l16  = lane & 15;
    const int hi   = lane >> 4;        // 0 or 1

    v8f acc[2][4] = {};

    // staging: each thread moves 16 bf16 (32B) of A and of B per K-step
    const int srow = tid >> 1;             // 0..127
    const int scol = (tid & 1) * 16;       // 0 or 16
    const __bf16* Ag = A  + (size_t)(bm + srow) * K + scol;
    const __bf16* Bg = Wb + (size_t)(bn + srow) * K + scol;

    // prologue: stage tile 0
    {
        v8bf a0 = *(const v8bf*)(Ag);
        v8bf a1 = *(const v8bf*)(Ag + 8);
        v8bf b0 = *(const v8bf*)(Bg);
        v8bf b1 = *(const v8bf*)(Bg + 8);
        *(v8bf*)&As[0][srow * LSTR + scol]     = a0;
        *(v8bf*)&As[0][srow * LSTR + scol + 8] = a1;
        *(v8bf*)&Bs[0][srow * LSTR + scol]     = b0;
        *(v8bf*)&Bs[0][srow * LSTR + scol + 8] = b1;
    }
    __syncthreads();

    int buf = 0;
    for (int kk = 0; kk < K; kk += 32) {
        const bool more = (kk + 32) < K;
        v8bf na0, na1, nb0, nb1;
        if (more) {
            na0 = *(const v8bf*)(Ag + kk + 32);
            na1 = *(const v8bf*)(Ag + kk + 40);
            nb0 = *(const v8bf*)(Bg + kk + 32);
            nb1 = *(const v8bf*)(Bg + kk + 40);
            __builtin_prefetch(Ag + kk + 64, 0, 1);
        }

        // ---- fragments from current buffer ----
        const __bf16* Ab = &As[buf][0];
        const __bf16* Bb = &Bs[buf][0];
        const int ar0 = (wm + l16) * LSTR;
        const int ar1 = (wm + 16 + l16) * LSTR;
        v16bf af0 = CAT16(*(const v8bf*)&Ab[ar0 + hi * 8],
                          *(const v8bf*)&Ab[ar0 + 16 + hi * 8]);
        v16bf af1 = CAT16(*(const v8bf*)&Ab[ar1 + hi * 8],
                          *(const v8bf*)&Ab[ar1 + 16 + hi * 8]);
        v16bf bfq[4];
        #pragma unroll
        for (int q = 0; q < 4; ++q) {
            const int br = (wn + q * 16 + l16) * LSTR + hi * 8;
            bfq[q] = CAT16(*(const v8bf*)&Bb[br], *(const v8bf*)&Bb[br + 16]);
        }
        #pragma unroll
        for (int q = 0; q < 4; ++q) {
            acc[0][q] = __builtin_amdgcn_wmma_f32_16x16x32_bf16(
                false, af0, false, bfq[q], (short)0, acc[0][q], false, false);
            acc[1][q] = __builtin_amdgcn_wmma_f32_16x16x32_bf16(
                false, af1, false, bfq[q], (short)0, acc[1][q], false, false);
        }

        // ---- stage next tile into the other buffer ----
        if (more) {
            const int nb = buf ^ 1;
            *(v8bf*)&As[nb][srow * LSTR + scol]     = na0;
            *(v8bf*)&As[nb][srow * LSTR + scol + 8] = na1;
            *(v8bf*)&Bs[nb][srow * LSTR + scol]     = nb0;
            *(v8bf*)&Bs[nb][srow * LSTR + scol + 8] = nb1;
        }
        __syncthreads();
        buf ^= 1;
    }

    // ---- epilogue: C/D layout -> VGPR j holds M = j + 8*hi, N = l16 ----
    #pragma unroll
    for (int i = 0; i < 2; ++i) {
        #pragma unroll
        for (int q = 0; q < 4; ++q) {
            const int c = bn + wn + q * 16 + l16;
            const float bb = bias ? bias[c] : 0.f;
            #pragma unroll
            for (int j = 0; j < 8; ++j) {
                const int r = bm + wm + i * 16 + hi * 8 + j;
                const size_t o = (size_t)r * N + c;
                float v = acc[i][q][j] + bb;
                if (mode == 0)      C[o] = v;
                else if (mode == 1) C[o] = fmaxf(v, 0.f);
                else if (mode == 2) C[o] += fmaxf(v, 0.f);
                else                C[o] = R[o] + fmaxf(v, 0.f);
            }
        }
    }
}

// ---------------------------------------------------------------------------
// Convert a weight matrix f32 -> bf16 once per GEMM (removes per-block
// redundant conversion and halves B-operand traffic).
// ---------------------------------------------------------------------------
__global__ __launch_bounds__(256)
void wconv_kernel(const float* __restrict__ W, __bf16* __restrict__ Wb)
{
    size_t e = (size_t)blockIdx.x * 256 + threadIdx.x;    // units of 4 elems
    v4f w = *(const v4f*)(W + e * 4);
    __bf16* d = Wb + e * 4;
    d[0] = (__bf16)w[0]; d[1] = (__bf16)w[1];
    d[2] = (__bf16)w[2]; d[3] = (__bf16)w[3];
}

// ---------------------------------------------------------------------------
// Attention: one block per (b,t). Gram-matrix formulation of the three
// l2-normalized score blocks. Writes pre-LN spatial (agg + x) to P and the
// softmax probs to attn_out. Dynamic LDS (~118 KB).
// ---------------------------------------------------------------------------
__global__ __launch_bounds__(256)
void attn_kernel(const float* __restrict__ node, const float* __restrict__ edge,
                 float* __restrict__ P, float* __restrict__ attn_out)
{
    extern __shared__ float sm[];
    float* xs   = sm;                 // 50*512
    float* sc   = xs + NN * HD;       // 50*52   (Gram -> scores -> probs)
    float* ev   = sc + NN * 52;       // 3*512   edge vectors
    float* pdot = ev + 3 * HD;        // 3*50    x_i . e_k
    float* ee   = pdot + 152;         // 3       ||e_k||^2
    float* nq   = ee + 4;             // 50      ||x_i||
    float* nk0  = nq + 64;            // 50      ||x_j + e0||
    float* nk1  = nk0 + 64;           // 50      ||x_j + e1||
    float* nkva = nk1 + 64;           // 1       ||x_49 + e2||

    const int bt  = blockIdx.x;
    const int tid = threadIdx.x;
    const float* xg = node + (size_t)bt * NN * HD;

    for (int u = tid; u < NN * HD / 4; u += 256)
        ((v4f*)xs)[u] = ((const v4f*)xg)[u];
    for (int u = tid; u < 3 * HD / 4; u += 256)
        ((v4f*)ev)[u] = ((const v4f*)edge)[u];
    __syncthreads();

    // x_i . e_k  and  ||e_k||^2
    for (int u = tid; u < 153; u += 256) {
        if (u < 150) {
            int k = u / 50, i = u % 50;
            const v4f* a = (const v4f*)(xs + i * HD);
            const v4f* b = (const v4f*)(ev + k * HD);
            float s = 0.f;
            for (int h = 0; h < HD / 4; ++h) {
                v4f x = a[h], y = b[h];
                s += x[0]*y[0] + x[1]*y[1] + x[2]*y[2] + x[3]*y[3];
            }
            pdot[k * 50 + i] = s;
        } else {
            int k = u - 150;
            const v4f* b = (const v4f*)(ev + k * HD);
            float s = 0.f;
            for (int h = 0; h < HD / 4; ++h) {
                v4f y = b[h];
                s += y[0]*y[0] + y[1]*y[1] + y[2]*y[2] + y[3]*y[3];
            }
            ee[k] = s;
        }
    }
    // Gram matrix G = x x^T
    for (int u = tid; u < NN * NN; u += 256) {
        int i = u / NN, j = u % NN;
        const v4f* a = (const v4f*)(xs + i * HD);
        const v4f* b = (const v4f*)(xs + j * HD);
        float s = 0.f;
        for (int h = 0; h < HD / 4; ++h) {
            v4f x = a[h], y = b[h];
            s += x[0]*y[0] + x[1]*y[1] + x[2]*y[2] + x[3]*y[3];
        }
        sc[i * 52 + j] = s;
    }
    __syncthreads();

    // norms (l2norm eps = 1e-12)
    if (tid < NN) {
        int i = tid;
        float g = sc[i * 52 + i];
        nq[i]  = fmaxf(sqrtf(fmaxf(g, 0.f)), 1e-12f);
        nk0[i] = fmaxf(sqrtf(fmaxf(g + 2.f * pdot[0 * 50 + i] + ee[0], 0.f)), 1e-12f);
        nk1[i] = fmaxf(sqrtf(fmaxf(g + 2.f * pdot[1 * 50 + i] + ee[1], 0.f)), 1e-12f);
        if (i == NN - 1)
            nkva[0] = fmaxf(sqrtf(fmaxf(g + 2.f * pdot[2 * 50 + 49] + ee[2], 0.f)), 1e-12f);
    }
    __syncthreads();

    // scores (in-place over Gram, pointwise)
    const float NEG = -1e9f;
    for (int u = tid; u < NN * NN; u += 256) {
        int i = u / NN, j = u % NN;
        float g = sc[i * 52 + j];
        float s;
        if (i == j) {
            s = NEG;
        } else if (j == 49) {                       // s_va column
            s = (g + pdot[2 * 50 + i]) / (nq[i] * nkva[0]);
        } else if (i == 49) {                       // s_av row
            s = (g + pdot[1 * 50 + 49]) / (nq[49] * nk1[j]);
        } else {                                    // s_vv with 3x3 window mask
            int qr = i / 7, qc = i % 7, kr = j / 7, kc = j % 7;
            int r0 = qr - 1; if (r0 < 0) r0 = 0;
            int c0 = qc - 1; if (c0 < 0) c0 = 0; if (c0 > 4) c0 = 4;
            bool ok = (kr >= r0) && (kr <= r0 + 2) && (kc >= c0) && (kc <= c0 + 2);
            s = ok ? (g + pdot[0 * 50 + i]) / (nq[i] * nk0[j]) : NEG;
        }
        sc[i * 52 + j] = s;
    }
    __syncthreads();

    // row softmax
    if (tid < NN) {
        float m = -3.4e38f;
        for (int j = 0; j < NN; ++j) m = fmaxf(m, sc[tid * 52 + j]);
        float sum = 0.f;
        for (int j = 0; j < NN; ++j) {
            float e = __expf(sc[tid * 52 + j] - m);
            sc[tid * 52 + j] = e;
            sum += e;
        }
        float inv = 1.f / sum;
        for (int j = 0; j < NN; ++j) sc[tid * 52 + j] *= inv;
    }
    __syncthreads();

    // export probs
    for (int u = tid; u < NN * NN; u += 256)
        attn_out[(size_t)bt * NN * NN + u] = sc[(u / NN) * 52 + (u % NN)];

    // aggregation + residual:  P[i] = x[i] + sum_j<49 a[i][j]*x[j]
    //                          + (1-a[i][49])*e_sel + a[i][49]*(x[49]+e2)
    for (int u = tid; u < NN * (HD / 4); u += 256) {
        int i  = u / (HD / 4);
        int hb = (u % (HD / 4)) * 4;
        float a49 = sc[i * 52 + 49];
        float rs  = 1.f - a49;                       // sum of a[i][0..48]
        const float* esel = (i < 49) ? (ev + hb) : (ev + HD + hb);
        float s0 = 0.f, s1 = 0.f, s2 = 0.f, s3 = 0.f;
        for (int j = 0; j < 49; ++j) {
            float aij = sc[i * 52 + j];
            const float* xj = xs + j * HD + hb;
            s0 += aij * xj[0]; s1 += aij * xj[1];
            s2 += aij * xj[2]; s3 += aij * xj[3];
        }
        const float* x49 = xs + 49 * HD + hb;
        const float* e2  = ev + 2 * HD + hb;
        const float* xi  = xs + i * HD + hb;
        v4f o;
        o[0] = xi[0] + s0 + rs * esel[0] + a49 * (x49[0] + e2[0]);
        o[1] = xi[1] + s1 + rs * esel[1] + a49 * (x49[1] + e2[1]);
        o[2] = xi[2] + s2 + rs * esel[2] + a49 * (x49[2] + e2[2]);
        o[3] = xi[3] + s3 + rs * esel[3] + a49 * (x49[3] + e2[3]);
        *(v4f*)(P + ((size_t)bt * NN + i) * HD + hb) = o;
    }
}

// ---------------------------------------------------------------------------
// Wave-per-row LayerNorm. Optional gated add (only when t>0), optional relu,
// optional fp32 and/or bf16 outputs.
// ---------------------------------------------------------------------------
__global__ __launch_bounds__(256)
void ln_kernel(const float* __restrict__ X, const float* __restrict__ Add,
               const float* __restrict__ g, const float* __restrict__ b,
               float* __restrict__ outF, __bf16* __restrict__ outB, int doRelu)
{
    const int row  = blockIdx.x * 8 + (threadIdx.x >> 5);
    const int lane = threadIdx.x & 31;
    const int t    = (row / NN) & (TQ - 1);
    const float* xr = X + (size_t)row * HD;
    const float* ar = Add + (size_t)row * HD;
    const bool useAdd = (Add != nullptr) && (t > 0);

    float v[16];
    float s = 0.f, s2 = 0.f;
    #pragma unroll
    for (int u = 0; u < 16; ++u) {
        float val = xr[lane + u * 32];
        if (useAdd) val += ar[lane + u * 32];
        v[u] = val; s += val; s2 += val * val;
    }
    for (int off = 16; off > 0; off >>= 1) {
        s  += __shfl_xor(s,  off, 32);
        s2 += __shfl_xor(s2, off, 32);
    }
    const float mean = s * (1.f / HD);
    const float var  = s2 * (1.f / HD) - mean * mean;
    const float inv  = rsqrtf(var + 1e-5f);
    #pragma unroll
    for (int u = 0; u < 16; ++u) {
        int c = lane + u * 32;
        float y = (v[u] - mean) * inv * g[c] + b[c];
        if (doRelu) y = fmaxf(y, 0.f);
        if (outF) outF[(size_t)row * HD + c] = y;
        if (outB) outB[(size_t)row * HD + c] = (__bf16)y;
    }
}

// ---------------------------------------------------------------------------
// Temporal cosine gating: wave per row. A1 = cos * prev, A2 = (1-cos) * cur
// (bf16 GEMM operands); cos written for rows t>=1; t==0 rows zeroed.
// ---------------------------------------------------------------------------
__global__ __launch_bounds__(256)
void cos_kernel(const float* __restrict__ S, __bf16* __restrict__ A1,
                __bf16* __restrict__ A2, float* __restrict__ cosOut)
{
    const int row  = blockIdx.x * 8 + (threadIdx.x >> 5);
    const int lane = threadIdx.x & 31;
    const int i    = row % NN;
    const int t    = (row / NN) & (TQ - 1);
    const int bb   = row / (NN * TQ);
    __bf16* a1 = A1 + (size_t)row * HD;
    __bf16* a2 = A2 + (size_t)row * HD;

    if (t == 0) {
        #pragma unroll
        for (int u = 0; u < 16; ++u) {
            a1[lane + u * 32] = (__bf16)0.f;
            a2[lane + u * 32] = (__bf16)0.f;
        }
        return;
    }
    const float* cur = S + (size_t)row * HD;
    const float* prv = cur - (size_t)NN * HD;
    float cv[16], pv[16];
    float d = 0.f, na = 0.f, nb = 0.f;
    #pragma unroll
    for (int u = 0; u < 16; ++u) {
        float c = cur[lane + u * 32], p = prv[lane + u * 32];
        cv[u] = c; pv[u] = p;
        d += c * p; na += c * c; nb += p * p;
    }
    for (int off = 16; off > 0; off >>= 1) {
        d  += __shfl_xor(d,  off, 32);
        na += __shfl_xor(na, off, 32);
        nb += __shfl_xor(nb, off, 32);
    }
    const float cosv = d / (fmaxf(sqrtf(na), 1e-8f) * fmaxf(sqrtf(nb), 1e-8f));
    if (lane == 0)
        cosOut[((size_t)bb * (TQ - 1) + (t - 1)) * NN + i] = cosv;
    const float om = 1.f - cosv;
    #pragma unroll
    for (int u = 0; u < 16; ++u) {
        a1[lane + u * 32] = (__bf16)(cosv * pv[u]);
        a2[lane + u * 32] = (__bf16)(om * cv[u]);
    }
}

// ---------------------------------------------------------------------------
// Pack concat(visual, audio) into bf16 X0
// ---------------------------------------------------------------------------
__global__ __launch_bounds__(256)
void pack_kernel(const float* __restrict__ vis, const float* __restrict__ aud,
                 __bf16* __restrict__ X)
{
    size_t e = (size_t)blockIdx.x * 256 + threadIdx.x;   // units of 4 elems
    if (e >= (size_t)MROWS * (HD / 4)) return;
    size_t r = e / (HD / 4);
    int hb   = (int)(e % (HD / 4)) * 4;
    int i    = (int)(r % NN);
    size_t bt = r / NN;
    const float* src = (i < 49) ? (vis + ((bt * 49 + i) * HD + hb))
                                : (aud + (bt * HD + hb));
    v4f w = *(const v4f*)src;
    __bf16* dst = X + r * HD + hb;
    dst[0] = (__bf16)w[0]; dst[1] = (__bf16)w[1];
    dst[2] = (__bf16)w[2]; dst[3] = (__bf16)w[3];
}

__global__ __launch_bounds__(256)
void tobf_kernel(const float* __restrict__ X, __bf16* __restrict__ Y)
{
    size_t e = (size_t)blockIdx.x * 256 + threadIdx.x;
    if (e >= (size_t)MROWS * (HD / 4)) return;
    v4f w = *(const v4f*)(X + e * 4);
    __bf16* dst = Y + e * 4;
    dst[0] = (__bf16)w[0]; dst[1] = (__bf16)w[1];
    dst[2] = (__bf16)w[2]; dst[3] = (__bf16)w[3];
}

// ---------------------------------------------------------------------------
// Launch
// ---------------------------------------------------------------------------
extern "C" void kernel_launch(void* const* d_in, const int* in_sizes, int n_in,
                              void* d_out, int out_size, void* d_ws, size_t ws_size,
                              hipStream_t stream)
{
    const float* vis   = (const float*)d_in[0];
    const float* aud   = (const float*)d_in[1];
    const float* in_W  = (const float*)d_in[2];
    const float* in_b  = (const float*)d_in[3];
    const float* out_W = (const float*)d_in[4];
    const float* out_b = (const float*)d_in[5];

    float* outp   = (float*)d_out;                         // [MROWS, 512]
    float* attn   = outp + (size_t)MROWS * HD;             // [1024, 50, 50]
    float* cosout = attn + (size_t)BQ * TQ * NN * NN;      // [16, 63, 50]

    float*  node = (float*)d_ws;                            // f32 MROWS*512
    float*  S    = node + (size_t)MROWS * HD;               // f32
    float*  P    = S    + (size_t)MROWS * HD;               // f32
    __bf16* A1   = (__bf16*)(P + (size_t)MROWS * HD);       // bf16
    __bf16* A2   = A1 + (size_t)MROWS * HD;                 // bf16
    __bf16* Wb   = A2 + (size_t)MROWS * HD;                 // bf16 512x512

    const dim3 gemmGrid(HD / 128, MROWS / 128);             // (4, 400)
    const int  rowBlocks  = MROWS / 8;                      // wave-per-row grids
    const int  elemBlocks = (int)(((size_t)MROWS * (HD / 4) + 255) / 256);
    const int  wconvBlocks = (HD * HD / 4) / 256;           // 256
    const size_t ATT_SMEM = (size_t)(NN * HD + NN * 52 + 3 * HD + 152 + 4 + 3 * 64 + 4) * 4;

    // input projection: node = concat(vis,aud) @ in_W^T + in_b
    pack_kernel<<<elemBlocks, 256, 0, stream>>>(vis, aud, A1);
    wconv_kernel<<<wconvBlocks, 256, 0, stream>>>(in_W, Wb);
    gemm_bf16<<<gemmGrid, 256, 0, stream>>>(A1, Wb, in_b, nullptr, node,
                                            MROWS, HD, HD, 0);

    for (int layer = 0; layer < 2; ++layer) {
        const float* edge = (const float*)d_in[6 + layer * 8 + 0];
        const float* ln_g = (const float*)d_in[6 + layer * 8 + 1];
        const float* ln_b = (const float*)d_in[6 + layer * 8 + 2];
        const float* Wc   = (const float*)d_in[6 + layer * 8 + 3];
        const float* bc   = (const float*)d_in[6 + layer * 8 + 4];
        const float* Ws   = (const float*)d_in[6 + layer * 8 + 5];
        const float* bs   = (const float*)d_in[6 + layer * 8 + 6];
        const float* Wo   = (const float*)d_in[6 + layer * 8 + 7];

        // P = agg + x ; attn probs exported
        attn_kernel<<<BQ * TQ, 256, ATT_SMEM, stream>>>(node, edge, P, attn);
        // S = relu(LN(P))
        ln_kernel<<<rowBlocks, 256, 0, stream>>>(P, nullptr, ln_g, ln_b,
                                                 S, nullptr, 1);
        // A1 = cos*prev, A2 = (1-cos)*cur  (bf16), cos exported
        cos_kernel<<<rowBlocks, 256, 0, stream>>>(S, A1, A2, cosout);
        // P = relu(A1 @ Wc^T + bc) ; P += relu(A2 @ Ws^T + bs)
        wconv_kernel<<<wconvBlocks, 256, 0, stream>>>(Wc, Wb);
        gemm_bf16<<<gemmGrid, 256, 0, stream>>>(A1, Wb, bc, nullptr, P,
                                                MROWS, HD, HD, 1);
        wconv_kernel<<<wconvBlocks, 256, 0, stream>>>(Ws, Wb);
        gemm_bf16<<<gemmGrid, 256, 0, stream>>>(A2, Wb, bs, nullptr, P,
                                                MROWS, HD, HD, 2);
        // A1 = bf16( LN( S + (t>0 ? P : 0) ) )
        ln_kernel<<<rowBlocks, 256, 0, stream>>>(S, P, ln_g, ln_b,
                                                 nullptr, A1, 0);
        // node = node + relu(A1 @ Wo^T)
        wconv_kernel<<<wconvBlocks, 256, 0, stream>>>(Wo, Wb);
        gemm_bf16<<<gemmGrid, 256, 0, stream>>>(A1, Wb, nullptr, node, node,
                                                MROWS, HD, HD, 3);
    }

    // out = node @ out_W^T + out_b
    tobf_kernel<<<elemBlocks, 256, 0, stream>>>(node, A1);
    wconv_kernel<<<wconvBlocks, 256, 0, stream>>>(out_W, Wb);
    gemm_bf16<<<gemmGrid, 256, 0, stream>>>(A1, Wb, out_b, nullptr, outp,
                                            MROWS, HD, HD, 0);
}
